// clusterLayer_18957985644844
// MI455X (gfx1250) — compile-verified, matrix operand order checked by reading, not run on hardware
//
#include <hip/hip_runtime.h>

// q = rownorm( 1 / (1 + ||x||^2 + ||c||^2 - 2 x.c) ),  alpha = 1
// N=524288, K=256, D=256.  bf16 WMMA matmul, fp32 accumulate, fp32 x2/c2.
// Fast path: clusters pre-packed to bf16 WMMA fragments in d_ws (L2-resident).

#define DDIM 256
#define KCL  256

typedef __attribute__((ext_vector_type(16))) __bf16 v16bf;
typedef __attribute__((ext_vector_type(8)))  float  v8f;

union BFPack {
    unsigned u[8];
    v16bf    v;
};

// one-op bf16 pair pack: {hi[31:16], lo[31:16]} via v_perm_b32
__device__ __forceinline__ unsigned pack2(float lo, float hi) {
    return __builtin_amdgcn_perm(__builtin_bit_cast(unsigned, hi),
                                 __builtin_bit_cast(unsigned, lo),
                                 0x07060302u);
}

// load 16 f32 in WMMA-fragment pattern, return sumsq + packed fragment
__device__ __forceinline__ float load_pack_frag(const float4* base, int f4,
                                                BFPack& P) {
    float4 f0 = base[f4], f1 = base[f4 + 1], f2 = base[f4 + 4], f3 = base[f4 + 5];
    float s = f0.x*f0.x + f0.y*f0.y + f0.z*f0.z + f0.w*f0.w
            + f1.x*f1.x + f1.y*f1.y + f1.z*f1.z + f1.w*f1.w
            + f2.x*f2.x + f2.y*f2.y + f2.z*f2.z + f2.w*f2.w
            + f3.x*f3.x + f3.y*f3.y + f3.z*f3.z + f3.w*f3.w;
    P.u[0] = pack2(f0.x, f0.y);
    P.u[1] = pack2(f0.z, f0.w);
    P.u[2] = pack2(f1.x, f1.y);
    P.u[3] = pack2(f1.z, f1.w);
    P.u[4] = pack2(f2.x, f2.y);
    P.u[5] = pack2(f2.z, f2.w);
    P.u[6] = pack2(f3.x, f3.y);
    P.u[7] = pack2(f3.z, f3.w);
    return s;
}

// shared epilogue: Student-t transform, block row-normalize, store
__device__ __forceinline__ void finish_store(const v8f& acc0, const v8f& acc1,
                                             float c20, float c21,
                                             const float* x2s, float* rowsum,
                                             int half, int l16, int n0, int m0,
                                             float* __restrict__ out) {
    float q0[8], q1[8];
#pragma unroll
    for (int i = 0; i < 8; ++i) {
        const int m = i + 8 * half;          // output row (C-matrix layout)
        const float xx = x2s[m];
        const float d0 = xx + c20 - 2.0f * acc0[i];
        const float d1 = xx + c21 - 2.0f * acc1[i];
        q0[i] = __builtin_amdgcn_rcpf(1.0f + d0);
        q1[i] = __builtin_amdgcn_rcpf(1.0f + d1);
        float r = q0[i] + q1[i];
        r += __shfl_xor(r, 1);
        r += __shfl_xor(r, 2);
        r += __shfl_xor(r, 4);
        r += __shfl_xor(r, 8);
        if (l16 == 0) atomicAdd(&rowsum[m], r);   // ds_add_f32
    }
    __syncthreads();
#pragma unroll
    for (int i = 0; i < 8; ++i) {
        const int m = i + 8 * half;
        const float inv = __builtin_amdgcn_rcpf(rowsum[m]);
        const size_t row = (size_t)(m0 + m) * KCL;
        out[row + n0 + l16]      = q0[i] * inv;
        out[row + n0 + 16 + l16] = q1[i] * inv;
    }
}

// ---- pre-pass: pack clusters into bf16 WMMA B-fragments + c2 ----
// grid 16 x 256: block ct packs column tile ct (16 clusters), all 8 K-steps.
__global__ __launch_bounds__(256)
void pack_b_kernel(const float* __restrict__ clus,
                   BFPack* __restrict__ bfrag,
                   float* __restrict__ c2g) {
    __shared__ float c2part[16];
    const int tid  = threadIdx.x;
    const int lane = tid & 31;
    const int l16  = lane & 15;
    const int kh   = (lane >> 4) * 8;
    const int j    = tid >> 5;               // K-step 0..7
    const int ct   = blockIdx.x;             // column tile 0..15

    if (tid < 16) c2part[tid] = 0.f;
    __syncthreads();

    const float4* b =
        reinterpret_cast<const float4*>(clus + (size_t)(ct * 16 + l16) * DDIM);
    BFPack P;
    float s = load_pack_frag(b, (j * 32 + kh) >> 2, P);
    bfrag[(ct * 8 + j) * 32 + lane] = P;
    atomicAdd(&c2part[l16], s);
    __syncthreads();
    if (tid < 16) c2g[ct * 16 + tid] = c2part[tid];
}

// ---- fast main kernel: B fragments pre-packed in global scratch ----
__global__ __launch_bounds__(256)
void cluster_q_fast(const float* __restrict__ ctx,
                    const BFPack* __restrict__ bfrag,
                    const float* __restrict__ c2g,
                    float* __restrict__ out) {
    __shared__ BFPack aLds[256];             // 8 KB A fragments
    __shared__ float  x2s[16];
    __shared__ float  rowsum[16];

    const int tid  = threadIdx.x;
    const int lane = tid & 31;
    const int wave = tid >> 5;
    const int half = lane >> 4;
    const int l16  = lane & 15;
    const int kh   = half * 8;
    const int m0   = blockIdx.x * 16;

    if (tid < 16) { x2s[tid] = 0.f; rowsum[tid] = 0.f; }
    __syncthreads();

    {   // stage A tile: thread t packs fragment (k-step t>>5, lane t&31)
        const int pj = tid >> 5;
        const float4* a =
            reinterpret_cast<const float4*>(ctx + (size_t)(m0 + l16) * DDIM);
        BFPack P;
        float s = load_pack_frag(a, (pj * 32 + kh) >> 2, P);
        aLds[tid] = P;                       // ds_store_b128 x2
        atomicAdd(&x2s[l16], s);
    }
    __syncthreads();

    const int n0 = wave * 32;
    const BFPack* bf0 = bfrag + (size_t)(2 * wave) * 256 + lane;
    const BFPack* bf1 = bf0 + 256;

    v8f acc0 = {};
    v8f acc1 = {};
#pragma unroll 4
    for (int j = 0; j < DDIM / 32; ++j) {
        BFPack A  = aLds[j * 32 + lane];     // ds_load_b128 x2
        BFPack B0 = bf0[j * 32];             // global_load_b128 x2 (bf16)
        BFPack B1 = bf1[j * 32];
        acc0 = __builtin_amdgcn_wmma_f32_16x16x32_bf16(
            false, A.v, false, B0.v, (short)0, acc0, false, false);
        acc1 = __builtin_amdgcn_wmma_f32_16x16x32_bf16(
            false, A.v, false, B1.v, (short)0, acc1, false, false);
    }

    finish_store(acc0, acc1, c2g[n0 + l16], c2g[n0 + 16 + l16],
                 x2s, rowsum, half, l16, n0, m0, out);
}

// ---- self-contained fallback (no scratch needed): pack B in-loop ----
__global__ __launch_bounds__(256)
void cluster_q_fb(const float* __restrict__ ctx,
                  const float* __restrict__ clus,
                  float* __restrict__ out) {
    __shared__ BFPack aLds[256];
    __shared__ float  x2s[16];
    __shared__ float  rowsum[16];

    const int tid  = threadIdx.x;
    const int lane = tid & 31;
    const int wave = tid >> 5;
    const int half = lane >> 4;
    const int l16  = lane & 15;
    const int kh   = half * 8;
    const int m0   = blockIdx.x * 16;

    if (tid < 16) { x2s[tid] = 0.f; rowsum[tid] = 0.f; }
    __syncthreads();

    {
        const int pj = tid >> 5;
        const float4* a =
            reinterpret_cast<const float4*>(ctx + (size_t)(m0 + l16) * DDIM);
        BFPack P;
        float s = load_pack_frag(a, (pj * 32 + kh) >> 2, P);
        aLds[tid] = P;
        atomicAdd(&x2s[l16], s);
    }
    __syncthreads();

    const int n0 = wave * 32;
    const float4* b0 =
        reinterpret_cast<const float4*>(clus + (size_t)(n0 + l16) * DDIM);
    const float4* b1 =
        reinterpret_cast<const float4*>(clus + (size_t)(n0 + 16 + l16) * DDIM);

    v8f acc0 = {};
    v8f acc1 = {};
    float c2p0 = 0.f, c2p1 = 0.f;
#pragma unroll 2
    for (int j = 0; j < DDIM / 32; ++j) {
        const int f4 = (j * 32 + kh) >> 2;
        BFPack A = aLds[j * 32 + lane];
        BFPack B0, B1;
        c2p0 += load_pack_frag(b0, f4, B0);
        c2p1 += load_pack_frag(b1, f4, B1);
        acc0 = __builtin_amdgcn_wmma_f32_16x16x32_bf16(
            false, A.v, false, B0.v, (short)0, acc0, false, false);
        acc1 = __builtin_amdgcn_wmma_f32_16x16x32_bf16(
            false, A.v, false, B1.v, (short)0, acc1, false, false);
    }
    c2p0 += __shfl_xor(c2p0, 16);            // combine complementary K halves
    c2p1 += __shfl_xor(c2p1, 16);

    finish_store(acc0, acc1, c2p0, c2p1, x2s, rowsum, half, l16, n0, m0, out);
}

extern "C" void kernel_launch(void* const* d_in, const int* in_sizes, int n_in,
                              void* d_out, int out_size, void* d_ws, size_t ws_size,
                              hipStream_t stream) {
    const float* ctx  = (const float*)d_in[0];   // (N, 256)
    const float* clus = (const float*)d_in[1];   // (256, 256)
    float* out = (float*)d_out;                  // (N, 256)

    const int N = in_sizes[0] / DDIM;            // 524288
    const size_t need = 4096 * sizeof(BFPack) + 256 * sizeof(float);

    if (ws_size >= need) {
        BFPack* bfrag = (BFPack*)d_ws;
        float*  c2w   = (float*)((char*)d_ws + 4096 * sizeof(BFPack));
        hipLaunchKernelGGL(pack_b_kernel, dim3(16), dim3(256), 0, stream,
                           clus, bfrag, c2w);
        hipLaunchKernelGGL(cluster_q_fast, dim3(N / 16), dim3(256), 0, stream,
                           ctx, bfrag, c2w, out);
    } else {
        hipLaunchKernelGGL(cluster_q_fb, dim3(N / 16), dim3(256), 0, stream,
                           ctx, clus, out);
    }
}
